// Arch7V3GraphEncoder_80187039416488
// MI455X (gfx1250) — matrix operand order repaired and checked by reading, not compiled, not run on hardware
//
#include <hip/hip_runtime.h>
#include <hip/hip_bf16.h>
#include <math.h>

typedef __attribute__((ext_vector_type(16))) _Float16 v16h;
typedef __attribute__((ext_vector_type(8)))  float    v8f;

#define HD 128
#define WPITCH 132   // padded LDS pitch (halves) to avoid bank conflicts

// ---------------- embed: h = (atom_emb[tok] + role_emb[is_root]) * valid ----------
__global__ __launch_bounds__(256) void embed_kernel(
    const int* __restrict__ x_tokens, const int* __restrict__ valid,
    const float* __restrict__ atom_emb, const float* __restrict__ role_emb,
    float* __restrict__ h, int SK)
{
    int gid = blockIdx.x * blockDim.x + threadIdx.x;
    if (gid >= SK * HD) return;
    int i = gid >> 7, c = gid & 127;
    int tok = x_tokens[i];
    int root = ((i & 7) == 0) ? 1 : 0;          // K = 8 nodes per subgraph
    float v = atom_emb[tok * HD + c] + role_emb[root * HD + c];
    h[gid] = valid[i] ? v : 0.0f;
}

// ---------------- edge scatter: aggr[dst] += relu(h[src] + bond_emb[tok]) --------
__global__ __launch_bounds__(256) void edge_kernel(
    const float* __restrict__ h, const int* __restrict__ src,
    const int* __restrict__ dst, const int* __restrict__ edge_tokens,
    const float* __restrict__ bond_emb, float* __restrict__ aggr, int E)
{
    int gid = blockIdx.x * blockDim.x + threadIdx.x;
    if (gid >= E * HD) return;
    int e = gid >> 7, c = gid & 127;
    float msg = h[(size_t)src[e] * HD + c] + bond_emb[edge_tokens[e] * HD + c];
    msg = msg > 0.0f ? msg : 0.0f;
    atomicAdd(&aggr[(size_t)dst[e] * HD + c], msg);
}

// ---------------- fused GIN MLP: h = mask(relu(z@W1+b1)@W2 + b2), z=(1+eps)h+aggr -
// block = 128 threads = 4 waves, 64 rows per block (16 rows per wave).
__global__ __launch_bounds__(128) void mlp_kernel(
    float* __restrict__ h, const float* __restrict__ aggr,
    const float* __restrict__ w1, const float* __restrict__ b1,
    const float* __restrict__ w2, const float* __restrict__ b2,
    const float* __restrict__ eps, const int* __restrict__ valid, int layer)
{
    __shared__ _Float16 wsh[HD * WPITCH];        // current weight matrix (N-major)
    __shared__ _Float16 stage[4][16 * WPITCH];   // per-wave intermediate 16x128 tile
    __shared__ float sb1[HD], sb2[HD];

    const int tid  = threadIdx.x;
    const int wave = tid >> 5, lane = tid & 31;
    const int mrow = lane & 15;                  // A: row-in-tile; B/D: col-in-tile
    const int hi   = lane >> 4;                  // lane-group selector
    const int kb   = hi * 8;                     // A-fragment K sub-base
    const int r0   = blockIdx.x * 64 + wave * 16;
    const float epsl = 1.0f + eps[layer];
    const float* W1 = w1 + (size_t)layer * HD * HD;
    const float* W2 = w2 + (size_t)layer * HD * HD;

    // --- load W1 transposed (f16) + biases into LDS ---
    for (int idx = tid; idx < HD * HD; idx += 128) {
        int k = idx >> 7, n = idx & 127;
        wsh[n * WPITCH + k] = (_Float16)W1[idx];
    }
    if (tid < HD) { sb1[tid] = b1[layer * HD + tid]; sb2[tid] = b2[layer * HD + tid]; }
    __syncthreads();

    // --- build A fragments for GEMM1: z = (1+eps)*h + aggr, per ISA 16-bit A layout ---
    const int arow = r0 + mrow;
    v16h a1[4];
#pragma unroll
    for (int kt = 0; kt < 4; ++kt) {
        int base = arow * HD + kt * 32 + kb;
#pragma unroll
        for (int j = 0; j < 8; ++j) {
            a1[kt][j]     = (_Float16)(epsl * h[base + j]      + aggr[base + j]);
            a1[kt][j + 8] = (_Float16)(epsl * h[base + 16 + j] + aggr[base + 16 + j]);
        }
    }

    // --- GEMM1: 8 N-tiles x 4 K-steps, bias + relu -> LDS stage (f16) ---
#pragma unroll
    for (int nt = 0; nt < 8; ++nt) {
        v8f acc = {};
#pragma unroll
        for (int kt = 0; kt < 4; ++kt) {
            const _Float16* bp = &wsh[(nt * 16 + mrow) * WPITCH + kt * 32 + hi * 8];
            v16h b;
#pragma unroll
            for (int j = 0; j < 8; ++j) { b[j] = bp[j]; b[j + 8] = bp[16 + j]; }
            acc = __builtin_amdgcn_wmma_f32_16x16x32_f16(
                false, a1[kt], false, b, (short)0, acc, false, false);
        }
        float bias = sb1[nt * 16 + mrow];
#pragma unroll
        for (int r = 0; r < 8; ++r) {
            float v = acc[r] + bias;
            v = v > 0.0f ? v : 0.0f;
            stage[wave][(r + 8 * hi) * WPITCH + nt * 16 + mrow] = (_Float16)v;
        }
    }
    __syncthreads();

    // --- swap weights: W2 into LDS ---
    for (int idx = tid; idx < HD * HD; idx += 128) {
        int k = idx >> 7, n = idx & 127;
        wsh[n * WPITCH + k] = (_Float16)W2[idx];
    }
    __syncthreads();

    // --- A fragments for GEMM2 from stage ---
    v16h a2[4];
#pragma unroll
    for (int kt = 0; kt < 4; ++kt) {
        const _Float16* sp = &stage[wave][mrow * WPITCH + kt * 32 + kb];
#pragma unroll
        for (int j = 0; j < 8; ++j) { a2[kt][j] = sp[j]; a2[kt][j + 8] = sp[16 + j]; }
    }

    // valid mask for the 8 rows this lane's D fragment covers
    float vf[8];
#pragma unroll
    for (int r = 0; r < 8; ++r) vf[r] = valid[r0 + r + 8 * hi] ? 1.0f : 0.0f;

    // --- GEMM2: bias + valid mask, write h in place ---
#pragma unroll
    for (int nt = 0; nt < 8; ++nt) {
        v8f acc = {};
#pragma unroll
        for (int kt = 0; kt < 4; ++kt) {
            const _Float16* bp = &wsh[(nt * 16 + mrow) * WPITCH + kt * 32 + hi * 8];
            v16h b;
#pragma unroll
            for (int j = 0; j < 8; ++j) { b[j] = bp[j]; b[j + 8] = bp[16 + j]; }
            acc = __builtin_amdgcn_wmma_f32_16x16x32_f16(
                false, a2[kt], false, b, (short)0, acc, false, false);
        }
        int n = nt * 16 + mrow;
        float bias = sb2[n];
#pragma unroll
        for (int r = 0; r < 8; ++r) {
            h[(size_t)(r0 + r + 8 * hi) * HD + n] = (acc[r] + bias) * vf[r];
        }
    }
}

// ---------------- scatter-mean flat nodes -> subgraphs (contiguous, no atomics) ---
__global__ __launch_bounds__(256) void submean_kernel(
    const float* __restrict__ h, const int* __restrict__ node_ids,
    float* __restrict__ hsub, int S)
{
    int gid = blockIdx.x * blockDim.x + threadIdx.x;
    if (gid >= S * HD) return;
    int s = gid >> 7, c = gid & 127;
    float sum = 0.0f, cnt = 0.0f;
#pragma unroll
    for (int k = 0; k < 8; ++k) {
        int idx = s * 8 + k;
        if (node_ids[idx] >= 0) { sum += h[(size_t)idx * HD + c]; cnt += 1.0f; }
    }
    hsub[gid] = sum / fmaxf(cnt, 1.0f);
}

// ---------------- HT softmax pooling + global add-pool per graph ------------------
__global__ __launch_bounds__(256) void pool_kernel(
    const float* __restrict__ hsub, const float* __restrict__ log_probs,
    const float* __restrict__ ht_alpha, const int* __restrict__ batch_graph,
    float* __restrict__ out, int n_total)
{
    int gid = blockIdx.x * blockDim.x + threadIdx.x;
    if (gid >= n_total * HD) return;
    int n = gid >> 7, c = gid & 127;
    float alpha = ht_alpha[0];
    float v[4];
#pragma unroll
    for (int m = 0; m < 4; ++m) {
        float lp = log_probs[n * 4 + m];
        lp = __builtin_isfinite(lp) ? lp : 0.0f;
        v[m] = -alpha * lp;
    }
    float mx = v[0];
#pragma unroll
    for (int m = 1; m < 4; ++m) mx = fmaxf(mx, v[m]);
    float e[4], s = 0.0f;
#pragma unroll
    for (int m = 0; m < 4; ++m) { e[m] = expf(v[m] - mx); s += e[m]; }
    float val = 0.0f;
#pragma unroll
    for (int m = 0; m < 4; ++m)
        val += (e[m] / s) * hsub[(size_t)(n * 4 + m) * HD + c];
    atomicAdd(&out[batch_graph[n] * HD + c], val);
}

extern "C" void kernel_launch(void* const* d_in, const int* in_sizes, int n_in,
                              void* d_out, int out_size, void* d_ws, size_t ws_size,
                              hipStream_t stream) {
    const int*   x_tokens    = (const int*)d_in[0];
    const int*   edge_tokens = (const int*)d_in[1];
    const int*   intra_ei    = (const int*)d_in[2];
    const int*   node_ids    = (const int*)d_in[3];
    const int*   valid       = (const int*)d_in[4];
    /* d_in[5] sub_batch: implicit (i / K), unused */
    const float* log_probs   = (const float*)d_in[6];
    const int*   batch_graph = (const int*)d_in[7];
    const float* atom_emb    = (const float*)d_in[8];
    const float* bond_emb    = (const float*)d_in[9];
    const float* role_emb    = (const float*)d_in[10];
    const float* eps         = (const float*)d_in[11];
    const float* w1          = (const float*)d_in[12];
    const float* b1          = (const float*)d_in[13];
    const float* w2          = (const float*)d_in[14];
    const float* b2          = (const float*)d_in[15];
    const float* ht_alpha    = (const float*)d_in[16];
    float* out = (float*)d_out;

    const int SK = in_sizes[0];       // 131072
    const int E  = in_sizes[1];       // 196608
    const int S  = in_sizes[6];       // 16384
    const int NT = in_sizes[7];       // 4096
    const int L  = in_sizes[11];      // 4

    float* h    = (float*)d_ws;                       // [SK,128] f32  (64 MB)
    float* aggr = h    + (size_t)SK * HD;             // [SK,128] f32  (64 MB)
    float* hsub = aggr + (size_t)SK * HD;             // [S,128]  f32  ( 8 MB)

    const int* srcp = intra_ei;       // intra_ei[0]
    const int* dstp = intra_ei + E;   // intra_ei[1]

    int thr = 256;
    embed_kernel<<<(SK * HD + thr - 1) / thr, thr, 0, stream>>>(
        x_tokens, valid, atom_emb, role_emb, h, SK);

    for (int l = 0; l < L; ++l) {
        hipMemsetAsync(aggr, 0, (size_t)SK * HD * sizeof(float), stream);
        edge_kernel<<<((size_t)E * HD + thr - 1) / thr, thr, 0, stream>>>(
            h, srcp, dstp, edge_tokens, bond_emb, aggr, E);
        mlp_kernel<<<SK / 64, 128, 0, stream>>>(
            h, aggr, w1, b1, w2, b2, eps, valid, l);
    }

    submean_kernel<<<(S * HD + thr - 1) / thr, thr, 0, stream>>>(h, node_ids, hsub, S);

    hipMemsetAsync(out, 0, (size_t)out_size * sizeof(float), stream);
    pool_kernel<<<(NT * HD + thr - 1) / thr, thr, 0, stream>>>(
        hsub, log_probs, ht_alpha, batch_graph, out, NT);
}
